// VLAModel_24395414241412
// MI455X (gfx1250) — matrix operand order repaired and checked by reading, not compiled, not run on hardware
//
#include <hip/hip_runtime.h>
#include <hip/hip_bf16.h>
#include <math.h>

typedef __attribute__((ext_vector_type(16))) _Float16 v16h;
typedef __attribute__((ext_vector_type(8)))  float    v8f;

#define BTOK   32768
#define NEXP   8

__device__ __forceinline__ float gelu_exact(float v) {
    return 0.5f * v * (1.0f + erff(v * 0.70710678118654752440f));
}

__device__ __forceinline__ float wave_red(float v) {
#pragma unroll
    for (int off = 16; off; off >>= 1) v += __shfl_xor(v, off, 32);
    return v;
}

__device__ __forceinline__ void async_copy_b128(unsigned lds_off, const void* gptr) {
    // GLOBAL_LOAD_ASYNC_TO_LDS_B128: copies 16 bytes from global memory into LDS, tracked by ASYNCcnt
    asm volatile("global_load_async_to_lds_b128 %0, %1, off"
                 :: "v"(lds_off), "v"((unsigned long long)(size_t)gptr)
                 : "memory");
}

__device__ __forceinline__ void wait_async0() {
#if __has_builtin(__builtin_amdgcn_s_wait_asynccnt)
    __builtin_amdgcn_s_wait_asynccnt(0);
#else
    asm volatile("s_wait_asynccnt 0x0" ::: "memory");
#endif
}

union HFrag { v16h v; unsigned u[8]; };

// ---------------------------------------------------------------------------
// Tiled WMMA GEMM: out[M,N] = epilogue(A[M,K] @ W[K,N] + bias)
// 128 threads (4 waves). Block tile 128x64, wave tile 32x64, K staged 32.
// A staged via async global->LDS b128; W staged as packed K-pair u32.
// ---------------------------------------------------------------------------
template<bool GELU, typename OT>
__global__ __launch_bounds__(128)
void gemm_wmma(const _Float16* __restrict__ A, const _Float16* __restrict__ W,
               const float* __restrict__ bias, OT* __restrict__ out,
               int M, int N, int K) {
    constexpr int KC = 32;
    __shared__ _Float16 As[128][KC + 8];   // row stride 80B (20 banks) -> conflict-free
    __shared__ unsigned Bs[64][17];        // Bs[n][v] = halves {K=2v, K=2v+1}; stride 17 banks

    const int tid  = threadIdx.x;
    const int lane = tid & 31;
    const int wave = tid >> 5;
    const int m0 = blockIdx.y * 128;
    const int n0 = blockIdx.x * 64;
    const int mrow = lane & 15;     // M-row within A frag / N-col within B frag
    const int half = lane >> 4;

    v8f c[2][4] = {};

    for (int k0 = 0; k0 < K; k0 += KC) {
        // ---- stage A tile 128xKC via async copy (4 x b128 per thread) ----
#pragma unroll
        for (int s = 0; s < 4; ++s) {
            int i = tid + s * 128;
            int r = i >> 2, cc = (i & 3) * 8;
            async_copy_b128((unsigned)(size_t)&As[r][cc],
                            A + (size_t)(m0 + r) * K + (k0 + cc));
        }
        // ---- stage W tile KCx64 as packed K-pairs (manual: needs repack) ----
        {
            int p  = tid >> 3;          // K-pair index 0..15
            int nb = tid & 7;           // 8-wide N chunk
            const _Float16* w0 = W + (size_t)(k0 + 2 * p)     * N + (n0 + nb * 8);
            const _Float16* w1 = W + (size_t)(k0 + 2 * p + 1) * N + (n0 + nb * 8);
            uint4 lo4 = *(const uint4*)w0;
            uint4 hi4 = *(const uint4*)w1;
            unsigned lo[4] = {lo4.x, lo4.y, lo4.z, lo4.w};
            unsigned hi[4] = {hi4.x, hi4.y, hi4.z, hi4.w};
#pragma unroll
            for (int q = 0; q < 4; ++q) {
                unsigned a = lo[q], b = hi[q];
                Bs[nb * 8 + 2 * q    ][p] = (a & 0xffffu) | (b << 16);
                Bs[nb * 8 + 2 * q + 1][p] = (a >> 16)     | (b & 0xffff0000u);
            }
        }
        wait_async0();
        __syncthreads();

        // ---- fragments ----
        HFrag a0, a1;
#pragma unroll
        for (int v = 0; v < 8; ++v) {
            int kk = 2 * v + 16 * half;
            a0.u[v] = *(const unsigned*)&As[wave * 32 +      mrow][kk];
            a1.u[v] = *(const unsigned*)&As[wave * 32 + 16 + mrow][kk];
        }
        HFrag b0, b1, b2, b3;
#pragma unroll
        for (int v = 0; v < 8; ++v) {
            int vv = v + 8 * half;      // K pair {2v+16*half, +1}
            b0.u[v] = Bs[ 0 + mrow][vv];
            b1.u[v] = Bs[16 + mrow][vv];
            b2.u[v] = Bs[32 + mrow][vv];
            b3.u[v] = Bs[48 + mrow][vv];
        }
        c[0][0] = __builtin_amdgcn_wmma_f32_16x16x32_f16(false, a0.v, false, b0.v, (short)0, c[0][0], false, false);
        c[0][1] = __builtin_amdgcn_wmma_f32_16x16x32_f16(false, a0.v, false, b1.v, (short)0, c[0][1], false, false);
        c[0][2] = __builtin_amdgcn_wmma_f32_16x16x32_f16(false, a0.v, false, b2.v, (short)0, c[0][2], false, false);
        c[0][3] = __builtin_amdgcn_wmma_f32_16x16x32_f16(false, a0.v, false, b3.v, (short)0, c[0][3], false, false);
        c[1][0] = __builtin_amdgcn_wmma_f32_16x16x32_f16(false, a1.v, false, b0.v, (short)0, c[1][0], false, false);
        c[1][1] = __builtin_amdgcn_wmma_f32_16x16x32_f16(false, a1.v, false, b1.v, (short)0, c[1][1], false, false);
        c[1][2] = __builtin_amdgcn_wmma_f32_16x16x32_f16(false, a1.v, false, b2.v, (short)0, c[1][2], false, false);
        c[1][3] = __builtin_amdgcn_wmma_f32_16x16x32_f16(false, a1.v, false, b3.v, (short)0, c[1][3], false, false);
        __syncthreads();
    }

    // Epilogue. C layout: lane n = lane&15, VGPR r -> M = r + 8*(lane>>4).
#pragma unroll
    for (int f = 0; f < 2; ++f) {
#pragma unroll
        for (int j = 0; j < 4; ++j) {
            int gc = n0 + j * 16 + mrow;
            float bv = bias[gc];
#pragma unroll
            for (int r = 0; r < 8; ++r) {
                int gm = m0 + wave * 32 + f * 16 + r + 8 * half;
                float v = c[f][j][r] + bv;
                if (GELU) v = gelu_exact(v);
                out[(size_t)gm * N + gc] = (OT)v;
            }
        }
    }
}

// ---------------------------------------------------------------------------
// Helpers
// ---------------------------------------------------------------------------
__global__ void concat_half(const float* __restrict__ vis, const float* __restrict__ lang,
                            const float* __restrict__ st, _Float16* __restrict__ xin) {
    int idx = blockIdx.x * 256 + threadIdx.x;
    int b = idx >> 9, c = idx & 511;
    float v = (c < 256) ? vis[b * 256 + c]
            : (c < 384) ? lang[b * 128 + (c - 256)]
                        : st[b * 128 + (c - 384)];
    xin[idx] = (_Float16)v;
}

__global__ void cvt_half(const float* __restrict__ s, _Float16* __restrict__ d, int n) {
    int i = blockIdx.x * 256 + threadIdx.x;
    if (i < n) d[i] = (_Float16)s[i];
}

// x = gelu(LN(t0)); writes f32 (for gate/residual) and f16 (for GEMMs)
__global__ __launch_bounds__(256)
void ln_gelu_kernel(const float* __restrict__ t0, const float* __restrict__ g,
                    const float* __restrict__ b, float* __restrict__ x,
                    _Float16* __restrict__ xh) {
    int tid = threadIdx.x, lane = tid & 31, wave = tid >> 5;
    size_t token = (size_t)blockIdx.x * 8 + wave;
    const float* row = t0 + token * 512;
    float v[16], s = 0.f, s2 = 0.f;
#pragma unroll
    for (int i = 0; i < 16; ++i) {
        v[i] = row[lane + 32 * i];
        s += v[i]; s2 += v[i] * v[i];
    }
    s = wave_red(s); s2 = wave_red(s2);
    float mu = s * (1.f / 512.f);
    float var = s2 * (1.f / 512.f) - mu * mu;
    float rstd = rsqrtf(var + 1e-5f);
#pragma unroll
    for (int i = 0; i < 16; ++i) {
        int c = lane + 32 * i;
        float y = (v[i] - mu) * rstd * g[c] + b[c];
        float o = gelu_exact(y);
        x[token * 512 + c]  = o;
        xh[token * 512 + c] = (_Float16)o;
    }
}

__global__ void zero_acc(float* acc) { if (threadIdx.x < 16) acc[threadIdx.x] = 0.f; }

// gate: logits = x @ Wg; softmax, top-2 -> combine[B,8]; lb-loss accumulators
__global__ __launch_bounds__(256)
void gate_kernel(const float* __restrict__ x, const float* __restrict__ Wg,
                 float* __restrict__ combine, float* __restrict__ acc) {
    __shared__ float sW[512 * 8];
    __shared__ float sCnt[8], sPs[8];
    int tid = threadIdx.x, lane = tid & 31, wave = tid >> 5;
    for (int i = tid; i < 4096; i += 256) sW[i] = Wg[i];
    if (tid < 8) { sCnt[tid] = 0.f; sPs[tid] = 0.f; }
    __syncthreads();

    size_t token = (size_t)blockIdx.x * 8 + wave;
    float l[8] = {0.f, 0.f, 0.f, 0.f, 0.f, 0.f, 0.f, 0.f};
    for (int k = lane; k < 512; k += 32) {
        float xv = x[token * 512 + k];
#pragma unroll
        for (int e = 0; e < 8; ++e) l[e] += xv * sW[k * 8 + e];
    }
#pragma unroll
    for (int e = 0; e < 8; ++e) l[e] = wave_red(l[e]);

    if (lane == 0) {
        float mx = l[0];
#pragma unroll
        for (int e = 1; e < 8; ++e) mx = fmaxf(mx, l[e]);
        float p[8], sum = 0.f;
#pragma unroll
        for (int e = 0; e < 8; ++e) { p[e] = expf(l[e] - mx); sum += p[e]; }
        float inv = 1.f / sum;
#pragma unroll
        for (int e = 0; e < 8; ++e) p[e] *= inv;
        int i1 = 0;
#pragma unroll
        for (int e = 1; e < 8; ++e) if (l[e] > l[i1]) i1 = e;
        int i2 = (i1 == 0) ? 1 : 0;
#pragma unroll
        for (int e = 0; e < 8; ++e) if (e != i1 && l[e] > l[i2]) i2 = e;
        float w1 = 1.f / (1.f + expf(l[i2] - l[i1]));
        float w2 = 1.f - w1;
#pragma unroll
        for (int e = 0; e < 8; ++e)
            combine[token * 8 + e] = (e == i1) ? w1 : ((e == i2) ? w2 : 0.f);
        atomicAdd(&sCnt[i1], 1.f);
        atomicAdd(&sCnt[i2], 1.f);
#pragma unroll
        for (int e = 0; e < 8; ++e) atomicAdd(&sPs[e], p[e]);
    }
    __syncthreads();
    if (tid < 8) {
        atomicAdd(&acc[tid], sCnt[tid]);
        atomicAdd(&acc[8 + tid], sPs[tid]);
    }
}

// out (+)= combine[:,e] * LN(x + h, ge, be); wave-per-token
__global__ __launch_bounds__(256)
void combine_ln_kernel(const float* __restrict__ x, const float* __restrict__ h,
                       const float* __restrict__ g, const float* __restrict__ b,
                       const float* __restrict__ combine, int e,
                       float* __restrict__ out, int first) {
    int tid = threadIdx.x, lane = tid & 31, wave = tid >> 5;
    size_t token = (size_t)blockIdx.x * 8 + wave;
    float w = combine[token * 8 + e];
    float v[16], s = 0.f, s2 = 0.f;
#pragma unroll
    for (int i = 0; i < 16; ++i) {
        size_t idx = token * 512 + lane + 32 * i;
        v[i] = x[idx] + h[idx];
        s += v[i]; s2 += v[i] * v[i];
    }
    s = wave_red(s); s2 = wave_red(s2);
    float mu = s * (1.f / 512.f);
    float var = s2 * (1.f / 512.f) - mu * mu;
    float rstd = rsqrtf(var + 1e-5f);
#pragma unroll
    for (int i = 0; i < 16; ++i) {
        int c = lane + 32 * i;
        float y = (v[i] - mu) * rstd * g[c] + b[c];
        size_t idx = token * 512 + c;
        float val = w * y;
        if (first) out[idx] = val; else out[idx] += val;
    }
}

__global__ void lb_final(const float* __restrict__ acc, float* __restrict__ out) {
    if (threadIdx.x == 0) {
        float lb = 0.f;
#pragma unroll
        for (int e = 0; e < 8; ++e)
            lb += (acc[e] / (32768.f * 2.f)) * (acc[8 + e] / 32768.f);
        out[0] = 8.f * lb;
    }
}

// ---------------------------------------------------------------------------
// Orchestration
// ---------------------------------------------------------------------------
extern "C" void kernel_launch(void* const* d_in, const int* in_sizes, int n_in,
                              void* d_out, int out_size, void* d_ws, size_t ws_size,
                              hipStream_t stream) {
    const float* vis   = (const float*)d_in[0];
    const float* lang  = (const float*)d_in[1];
    const float* state = (const float*)d_in[2];
    const float* Wf    = (const float*)d_in[3];
    const float* bf    = (const float*)d_in[4];
    const float* gf    = (const float*)d_in[5];
    const float* betaf = (const float*)d_in[6];
    const float* Wg    = (const float*)d_in[7];
    const float* W1    = (const float*)d_in[8];
    const float* b1    = (const float*)d_in[9];
    const float* W2    = (const float*)d_in[10];
    const float* b2    = (const float*)d_in[11];
    const float* W3    = (const float*)d_in[12];
    const float* b3    = (const float*)d_in[13];
    const float* ge    = (const float*)d_in[14];
    const float* be    = (const float*)d_in[15];

    float* out = (float*)d_out;

    char* base = (char*)d_ws;
    size_t off = 0;
    auto carve = [&](size_t bytes) -> void* {
        off = (off + 255) & ~(size_t)255;
        void* p = base + off;
        off += bytes;
        return p;
    };
    const size_t B = BTOK;
    _Float16* xin  = (_Float16*)carve(B * 512  * 2);
    _Float16* Wfh  = (_Float16*)carve(512 * 512 * 2);
    _Float16* W1h  = (_Float16*)carve((size_t)NEXP * 512 * 1024 * 2);
    _Float16* W2h  = (_Float16*)carve((size_t)NEXP * 1024 * 512 * 2);
    _Float16* W3h  = (_Float16*)carve((size_t)NEXP * 512 * 512 * 2);
    float*    t0   = (float*)   carve(B * 512  * 4);  // fusion pre-LN, reused as expert h3
    float*    x    = (float*)   carve(B * 512  * 4);
    _Float16* xh   = (_Float16*)carve(B * 512  * 2);
    _Float16* h1   = (_Float16*)carve(B * 1024 * 2);
    _Float16* h2   = (_Float16*)carve(B * 512  * 2);
    float*    comb = (float*)   carve(B * 8    * 4);
    float*    acc  = (float*)   carve(16 * 4);

    // 1. concat + f16 conversions
    concat_half<<<(B * 512) / 256, 256, 0, stream>>>(vis, lang, state, xin);
    {
        int n;
        n = 512 * 512;         cvt_half<<<(n + 255) / 256, 256, 0, stream>>>(Wf, Wfh, n);
        n = NEXP * 512 * 1024; cvt_half<<<(n + 255) / 256, 256, 0, stream>>>(W1, W1h, n);
        n = NEXP * 1024 * 512; cvt_half<<<(n + 255) / 256, 256, 0, stream>>>(W2, W2h, n);
        n = NEXP * 512 * 512;  cvt_half<<<(n + 255) / 256, 256, 0, stream>>>(W3, W3h, n);
    }

    // 2. fusion GEMM -> t0, then LN+GELU -> x (f32) + xh (f16)
    gemm_wmma<false, float>
        <<<dim3(512 / 64, B / 128), 128, 0, stream>>>(xin, Wfh, bf, t0, B, 512, 512);
    ln_gelu_kernel<<<B / 8, 256, 0, stream>>>(t0, gf, betaf, x, xh);

    // 3. gate
    zero_acc<<<1, 32, 0, stream>>>(acc);
    gate_kernel<<<B / 8, 256, 0, stream>>>(x, Wg, comb, acc);

    // 4. experts (dense compute, sparse combine — matches reference)
    for (int e = 0; e < NEXP; ++e) {
        gemm_wmma<true, _Float16>
            <<<dim3(1024 / 64, B / 128), 128, 0, stream>>>(
                xh, W1h + (size_t)e * 512 * 1024, b1 + (size_t)e * 1024, h1, B, 1024, 512);
        gemm_wmma<true, _Float16>
            <<<dim3(512 / 64, B / 128), 128, 0, stream>>>(
                h1, W2h + (size_t)e * 1024 * 512, b2 + (size_t)e * 512, h2, B, 512, 1024);
        gemm_wmma<false, float>
            <<<dim3(512 / 64, B / 128), 128, 0, stream>>>(
                h2, W3h + (size_t)e * 512 * 512, b3 + (size_t)e * 512, t0, B, 512, 512);
        combine_ln_kernel<<<B / 8, 256, 0, stream>>>(
            x, t0, ge + (size_t)e * 512, be + (size_t)e * 512, comb, e, out, e == 0);
    }

    // 5. load-balance loss scalar
    lb_final<<<1, 32, 0, stream>>>(acc, out + B * 512);
}
